// SparseGAT_7378753815341
// MI455X (gfx1250) — matrix operand order repaired and checked by reading, not compiled, not run on hardware
//
#include <hip/hip_runtime.h>
#include <hip/hip_bf16.h>

typedef __attribute__((ext_vector_type(16))) __bf16 v16bf;
typedef __attribute__((ext_vector_type(8)))  __bf16 v8bf;
typedef __attribute__((ext_vector_type(8)))  float  v8f;

namespace {
constexpr int kN  = 50000;   // nodes
constexpr int kE  = 800000;  // edges
constexpr int kFI = 256;     // input features
constexpr int kFO = 64;      // output features per head
constexpr int kH  = 8;       // heads
constexpr int TM  = 128;     // rows per GEMM block
constexpr int LDW = kFI + 8; // padded K-leading-dim for transposed W in LDS
constexpr float kNegSlope = 0.2f;

__device__ __forceinline__ unsigned encOrderedF32(float f) {
  unsigned b = __float_as_uint(f);
  return (b & 0x80000000u) ? ~b : (b | 0x80000000u);
}
__device__ __forceinline__ float decOrderedF32(unsigned e) {
  unsigned b = (e & 0x80000000u) ? (e ^ 0x80000000u) : ~e;
  return __uint_as_float(b);
}
} // namespace

// ---------------------------------------------------------------------------
// Dense per-head GEMM  t = x @ W[h]  (bf16 WMMA, f32 accumulate),
// fused with aw = t @ a[h] via wave32 shfl_xor reduction.
//
// Block: 256 threads = 8 waves; tile 128 (M) x 64 (N), full K = 256.
//  - A fragments: loaded directly from global x as two contiguous 8-f32 runs
//    per lane per k-step (ISA 7.12.2 A layout), converted to bf16 in regs.
//  - B fragments: W[h] staged once into LDS *transposed* (sWt[n][k], bf16,
//    +8 pad) so each fragment is two 16-byte runs (k = 8h.., 16+8h..) -> 2x ds_load_b128.
// ---------------------------------------------------------------------------
__global__ void __launch_bounds__(256)
gat_gemm_aw_kernel(const float* __restrict__ x,   // [N][FI]
                   const float* __restrict__ W,   // [H][FI][FO]
                   const float* __restrict__ a,   // [H][FO]
                   float* __restrict__ t,         // [N][FO]
                   float* __restrict__ aw,        // [N]
                   int head) {
  __shared__ __bf16 sWt[kFO * LDW];  // 64 x 264 bf16 = 33 KB

  const int tid   = threadIdx.x;
  const int wave  = tid >> 5;
  const int lane  = tid & 31;
  const int half  = lane >> 4;   // K-striping half (ISA 7.12.2)
  const int m     = lane & 15;
  const int mbase = blockIdx.x * TM;

  const float* Wh = W + (size_t)head * kFI * kFO;

  // ---- stage W[h] transposed into LDS: sWt[n][k] ----
  // thread -> (n = tid&63, kg = tid>>6); global reads coalesced along n.
  {
    const int n  = tid & (kFO - 1);
    const int kg = tid >> 6;             // 0..3, each covers 64 k values
#pragma unroll
    for (int j = 0; j < 8; ++j) {        // 8 x v8bf stores
      v8bf wv;
#pragma unroll
      for (int u = 0; u < 8; ++u) {
        const int k = kg * 64 + j * 8 + u;
        wv[u] = (__bf16)Wh[(size_t)k * kFO + n];
      }
      *reinterpret_cast<v8bf*>(&sWt[n * LDW + kg * 64 + j * 8]) = wv;
    }
  }
  __syncthreads();

  // ---- main WMMA loop over K ----
  const int arow = mbase + wave * 16 + m;
  const int arowC = (arow < kN) ? arow : (kN - 1);   // clamped for loads
  const float* xrow = x + (size_t)arowC * kFI;

  v8f acc[4] = {};   // 4 n-tiles of 16x16 f32 accumulators

  for (int ks = 0; ks < kFI / 32; ++ks) {
    // A fragment: rows contiguous in K -> two 8-f32 runs, 4x b128 loads
    const int o1 = ks * 32 + half * 8;
    const float4 A0 = *reinterpret_cast<const float4*>(xrow + o1);
    const float4 A1 = *reinterpret_cast<const float4*>(xrow + o1 + 4);
    const float4 A2 = *reinterpret_cast<const float4*>(xrow + o1 + 16);
    const float4 A3 = *reinterpret_cast<const float4*>(xrow + o1 + 20);
    v16bf af;
    af[0]  = (__bf16)A0.x; af[1]  = (__bf16)A0.y;
    af[2]  = (__bf16)A0.z; af[3]  = (__bf16)A0.w;
    af[4]  = (__bf16)A1.x; af[5]  = (__bf16)A1.y;
    af[6]  = (__bf16)A1.z; af[7]  = (__bf16)A1.w;
    af[8]  = (__bf16)A2.x; af[9]  = (__bf16)A2.y;
    af[10] = (__bf16)A2.z; af[11] = (__bf16)A2.w;
    af[12] = (__bf16)A3.x; af[13] = (__bf16)A3.y;
    af[14] = (__bf16)A3.z; af[15] = (__bf16)A3.w;

#pragma unroll
    for (int nt = 0; nt < 4; ++nt) {
      // B fragment: two 16-byte runs (k = 8*half.., 16+8*half..) of sWt row (nt*16+m)
      const int boff = (nt * 16 + m) * LDW + ks * 32 + half * 8;
      const v8bf blo = *reinterpret_cast<const v8bf*>(&sWt[boff]);
      const v8bf bhi = *reinterpret_cast<const v8bf*>(&sWt[boff + 16]);
      v16bf bf;
#pragma unroll
      for (int e = 0; e < 8; ++e) { bf[e] = blo[e]; bf[e + 8] = bhi[e]; }
      acc[nt] = __builtin_amdgcn_wmma_f32_16x16x32_bf16(
          false, af, false, bf, (short)0, acc[nt], false, false);
    }
  }

  // ---- epilogue: store t; fused aw = t . a[head] ----
  float aval[4];
#pragma unroll
  for (int nt = 0; nt < 4; ++nt) aval[nt] = a[head * kFO + nt * 16 + m];

#pragma unroll
  for (int r = 0; r < 8; ++r) {
    const int row = mbase + wave * 16 + half * 8 + r;  // D layout: M = 8*half + r
    float p = 0.f;
#pragma unroll
    for (int nt = 0; nt < 4; ++nt) {
      const float v = acc[nt][r];
      if (row < kN) t[(size_t)row * kFO + nt * 16 + m] = v;
      p += v * aval[nt];
    }
    // reduce across the 16 lanes of this half (offsets 1..8 stay in-half)
#pragma unroll
    for (int off = 1; off < 16; off <<= 1) p += __shfl_xor(p, off, 32);
    if (m == 0 && row < kN) aw[row] = p;
  }
}

// ---------------------------------------------------------------------------
// Sparse softmax + aggregation pipeline
// ---------------------------------------------------------------------------
__global__ void gat_zero_acc_kernel(float* __restrict__ acc) {
  const int i = blockIdx.x * blockDim.x + threadIdx.x;
  if (i < kN * kFO) acc[i] = 0.f;
}

__global__ void gat_init_head_kernel(unsigned* __restrict__ m_enc,
                                     float* __restrict__ den) {
  const int i = blockIdx.x * blockDim.x + threadIdx.x;
  if (i < kN) {
    m_enc[i] = 0x007FFFFFu;  // encOrderedF32(-inf)
    den[i] = 0.f;
  }
}

__global__ void gat_edge_max_kernel(const float* __restrict__ aw,
                                    const int* __restrict__ rows,
                                    const int* __restrict__ cols,
                                    unsigned* __restrict__ m_enc) {
  const int k = blockIdx.x * blockDim.x + threadIdx.x;
  if (k >= kE) return;
  const float s = aw[rows[k]] + aw[cols[k]];
  const float e = (s > 0.f) ? s : kNegSlope * s;
  atomicMax(&m_enc[rows[k]], encOrderedF32(e));
}

__global__ void gat_edge_exp_kernel(const float* __restrict__ aw,
                                    const int* __restrict__ rows,
                                    const int* __restrict__ cols,
                                    const unsigned* __restrict__ m_enc,
                                    float* __restrict__ ex,
                                    float* __restrict__ den) {
  const int k = blockIdx.x * blockDim.x + threadIdx.x;
  if (k >= kE) return;
  const int r = rows[k];
  const float s = aw[r] + aw[cols[k]];
  const float e = (s > 0.f) ? s : kNegSlope * s;
  const float v = __expf(e - decOrderedF32(m_enc[r]));
  ex[k] = v;
  atomicAdd(&den[r], v);
}

// 16 lanes per edge, 4 features per lane -> coalesced 64B per quarter-row.
__global__ void gat_aggregate_kernel(const float* __restrict__ t,
                                     const int* __restrict__ rows,
                                     const int* __restrict__ cols,
                                     const float* __restrict__ ex,
                                     const float* __restrict__ den,
                                     float* __restrict__ acc) {
  const long tid = (long)blockIdx.x * blockDim.x + threadIdx.x;
  const int k = (int)(tid >> 4);
  if (k >= kE) return;
  const int g = (int)(tid & 15);
  const int r = rows[k];
  const int c = cols[k];
  // hint at an upcoming gather row (t is L2-resident)
  if (k + 32 < kE) __builtin_prefetch(t + (size_t)cols[k + 32] * kFO, 0, 0);
  const float coef = ex[k] / den[r];
  const float* trow = t + (size_t)c * kFO;
  float* arow = acc + (size_t)r * kFO;
#pragma unroll
  for (int q = 0; q < 4; ++q) {
    const int f = g + 16 * q;
    atomicAdd(&arow[f], coef * trow[f]);
  }
}

__global__ void gat_final_kernel(const float* __restrict__ acc,
                                 const float* __restrict__ b,  // [H][FO]
                                 float* __restrict__ out) {
  const int i = blockIdx.x * blockDim.x + threadIdx.x;
  if (i >= kN * kFO) return;
  const int f = i & (kFO - 1);
  float bs = 0.f;
#pragma unroll
  for (int h = 0; h < kH; ++h) bs += b[h * kFO + f];
  const float v = (acc[i] + bs) * (1.f / kH);
  out[i] = v > 0.f ? v : 0.f;
}

// ---------------------------------------------------------------------------
extern "C" void kernel_launch(void* const* d_in, const int* in_sizes, int n_in,
                              void* d_out, int out_size, void* d_ws,
                              size_t ws_size, hipStream_t stream) {
  (void)in_sizes; (void)n_in; (void)out_size; (void)ws_size;
  const float* x    = (const float*)d_in[0];
  const float* W    = (const float*)d_in[1];
  const float* a    = (const float*)d_in[2];
  const float* b    = (const float*)d_in[3];
  const int*   rows = (const int*)d_in[4];
  const int*   cols = (const int*)d_in[5];
  float* out = (float*)d_out;

  char* ws = (char*)d_ws;
  size_t off = 0;
  auto alloc = [&](size_t bytes) -> void* {
    void* p = ws + off;
    off = (off + bytes + 255) & ~(size_t)255;
    return p;
  };
  float*    acc   = (float*)alloc(sizeof(float) * kN * kFO);  // 12.8 MB
  float*    t     = (float*)alloc(sizeof(float) * kN * kFO);  // 12.8 MB
  float*    aw    = (float*)alloc(sizeof(float) * kN);
  unsigned* m_enc = (unsigned*)alloc(sizeof(unsigned) * kN);
  float*    den   = (float*)alloc(sizeof(float) * kN);
  float*    ex    = (float*)alloc(sizeof(float) * kE);        // 3.2 MB

  const dim3 blk(256);
  gat_zero_acc_kernel<<<(kN * kFO + 255) / 256, blk, 0, stream>>>(acc);

  const int gemmBlocks = (kN + TM - 1) / TM;  // 391
  for (int h = 0; h < kH; ++h) {
    gat_init_head_kernel<<<(kN + 255) / 256, blk, 0, stream>>>(m_enc, den);
    gat_gemm_aw_kernel<<<gemmBlocks, blk, 0, stream>>>(x, W, a, t, aw, h);
    gat_edge_max_kernel<<<(kE + 255) / 256, blk, 0, stream>>>(aw, rows, cols,
                                                              m_enc);
    gat_edge_exp_kernel<<<(kE + 255) / 256, blk, 0, stream>>>(aw, rows, cols,
                                                              m_enc, ex, den);
    gat_aggregate_kernel<<<(kE * 16 + 255) / 256, blk, 0, stream>>>(
        t, rows, cols, ex, den, acc);
  }
  gat_final_kernel<<<(kN * kFO + 255) / 256, blk, 0, stream>>>(acc, b, out);
}
